// GRUNet_24421184045356
// MI455X (gfx1250) — matrix operand order repaired and checked by reading, not compiled, not run on hardware
//
#include <hip/hip_runtime.h>

// GRUNet: 2-layer GRU (H=64) + FC, B=2048, T=512, in=7.
// One wave32 per 16 batch rows; all weight B-fragments held in VGPRs across
// the 512-step scan; recurrent + input GEMMs via v_wmma_f32_16x16x32_f16.

#define T_SEQ 512
#define BATCH 2048
#define HID   64
#define GATES 192
#define ROWS_PER_WAVE 16

typedef __attribute__((ext_vector_type(16))) _Float16 v16h;
typedef __attribute__((ext_vector_type(8)))  _Float16 v8h;
typedef __attribute__((ext_vector_type(8)))  float    v8f;

__device__ __forceinline__ float fast_tanh(float x) {
#if __has_builtin(__builtin_amdgcn_tanhf)
  return __builtin_amdgcn_tanhf(x);
#else
  float r;
  asm volatile("v_tanh_f32 %0, %1" : "=v"(r) : "v"(x));
  return r;
#endif
}

__device__ __forceinline__ v8f wmma(v16h a, v16h b, v8f c) {
  return __builtin_amdgcn_wmma_f32_16x16x32_f16(false, a, false, b, (short)0, c,
                                                false, false);
}

// B-operand (32x16 f16) built once from global f32 weights (row-major 192 x K):
// lane holds column n = tile*16 + (lane&15); lanes 0-15 -> K 0..15,
// lanes 16-31 -> K 16..31 of 32-K chunk kc -> one contiguous 16-float row chunk.
template<int K>
__device__ __forceinline__ v16h load_bf32(const float* W, int tile, int kc, int lane) {
  int n    = tile * 16 + (lane & 15);
  int koff = kc * 32 + ((lane & 16) ? 16 : 0);
  const float* p = W + n * K + koff;
  v16h r;
#pragma unroll
  for (int d = 0; d < 16; ++d) r[d] = (_Float16)p[d];
  return r;
}

// Layer-0 W_ih (192 x 7) padded to K=32: only K 0..6 nonzero (lanes 0-15 only).
__device__ __forceinline__ v16h load_bih0(const float* W, int tile, int lane) {
  int n = tile * 16 + (lane & 15);
  v16h r = {};
  if ((lane & 16) == 0) {
#pragma unroll
    for (int d = 0; d < 7; ++d) r[d] = (_Float16)W[n * 7 + d];
  }
  return r;
}

// A-operand (16x32 f16) from a row-major [16 x 64] half buffer:
// lane holds row m = lane&15; halves {c0..c0+15}, c0=(lane>=16)?16:0.
__device__ __forceinline__ v16h load_a64(const _Float16* Hbuf, int kc, int lane) {
  int m  = lane & 15;
  int c0 = (lane & 16) ? 16 : 0;
  const _Float16* p = Hbuf + m * 64 + kc * 32 + c0;
  union { v16h v; v8h h[2]; } u;
  u.h[0] = *(const v8h*)(p);
  u.h[1] = *(const v8h*)(p + 8);
  return u.v;
}

// x_t A-fragment: K=32 chunk, only K 0..6 populated.
__device__ __forceinline__ v16h load_xa(const float* x, int row0, int t, int lane) {
  v16h a = {};
  if (lane < 16) {
    const float* xp = x + ((size_t)(row0 + lane) * T_SEQ + t) * 7;
#pragma unroll
    for (int d = 0; d < 7; ++d) a[d] = (_Float16)xp[d];
  }
  return a;
}

// h1 (f16 global, (B,T,64)) A-fragments for chunk kc.
__device__ __forceinline__ v16h load_ga(const _Float16* h1, int row0, int t,
                                        int kc, int lane) {
  int m  = lane & 15;
  int c0 = (lane & 16) ? 16 : 0;
  const _Float16* p = h1 + ((size_t)(row0 + m) * T_SEQ + t) * 64 + kc * 32 + c0;
  union { v16h v; v8h h[2]; } u;
  u.h[0] = *(const v8h*)(p);
  u.h[1] = *(const v8h*)(p + 8);
  return u.v;
}

__global__ __launch_bounds__(32)
void gru_layer0(const float* __restrict__ x,        // (B, T, 7)
                const float* __restrict__ W_ih,     // (192, 7)
                const float* __restrict__ W_hh,     // (192, 64)
                const float* __restrict__ b_ih,
                const float* __restrict__ b_hh,
                _Float16*    __restrict__ h1)       // (B, T, 64) f16
{
  __shared__ _Float16 sH[16 * 64];

  const int lane = threadIdx.x & 31;
  const int row0 = blockIdx.x * ROWS_PER_WAVE;
  const int col  = lane & 15;

  // Loop-invariant weight B-fragments held in VGPRs for the whole scan.
  v16h Bih[12], Bhh0[12], Bhh1[12];
#pragma unroll
  for (int j = 0; j < 12; ++j) {
    Bih[j]  = load_bih0(W_ih, j, lane);
    Bhh0[j] = load_bf32<64>(W_hh, j, 0, lane);
    Bhh1[j] = load_bf32<64>(W_hh, j, 1, lane);
  }

  float hbrz[8], bxn[4], bhn[4];
#pragma unroll
  for (int j = 0; j < 8; ++j)
    hbrz[j] = 0.5f * (b_ih[j * 16 + col] + b_hh[j * 16 + col]);
#pragma unroll
  for (int j = 0; j < 4; ++j) {
    bxn[j] = b_ih[128 + j * 16 + col];
    bhn[j] = b_hh[128 + j * 16 + col];
  }

  for (int i = lane; i < 16 * 64; i += 32) sH[i] = (_Float16)0.0f;
  asm volatile("s_wait_dscnt 0x0" ::: "memory");

  v8f hfrag[4];
#pragma unroll
  for (int j = 0; j < 4; ++j) hfrag[j] = {};

  v16h xa_n = load_xa(x, row0, 0, lane);   // software-pipelined x prefetch

  for (int t = 0; t < T_SEQ; ++t) {
    v16h xa = xa_n;
    int tn = (t + 1 < T_SEQ) ? t + 1 : t;
    xa_n = load_xa(x, row0, tn, lane);

    v16h ha0 = load_a64(sH, 0, lane);
    v16h ha1 = load_a64(sH, 1, lane);

    v8f rz[8], xn[4], hn[4];
#pragma unroll
    for (int j = 0; j < 8; ++j) {
      v8f acc = {};
      acc = wmma(xa,  Bih[j],  acc);
      acc = wmma(ha0, Bhh0[j], acc);
      acc = wmma(ha1, Bhh1[j], acc);
      rz[j] = acc;
    }
#pragma unroll
    for (int j = 0; j < 4; ++j) {
      v8f a1 = {}, a2 = {};
      a1 = wmma(xa,  Bih[8 + j],  a1);
      a2 = wmma(ha0, Bhh0[8 + j], a2);
      a2 = wmma(ha1, Bhh1[8 + j], a2);
      xn[j] = a1; hn[j] = a2;
    }

#pragma unroll
    for (int j = 0; j < 4; ++j) {
#pragma unroll
      for (int i = 0; i < 8; ++i) {
        float rv = fmaf(0.5f, fast_tanh(fmaf(0.5f, rz[j][i], hbrz[j])), 0.5f);
        float zv = fmaf(0.5f, fast_tanh(fmaf(0.5f, rz[4 + j][i], hbrz[4 + j])), 0.5f);
        float na = xn[j][i] + bxn[j];
        na = fmaf(rv, hn[j][i], na);
        na = fmaf(rv, bhn[j], na);
        float nv = fast_tanh(na);
        hfrag[j][i] = fmaf(zv, hfrag[j][i] - nv, nv);
      }
    }

    // D-layout -> row-major LDS (rows m = vgpr + 8*(lane>=16), col = tile*16 + lane&15)
    const int rbase = (lane & 16) ? 8 : 0;
#pragma unroll
    for (int j = 0; j < 4; ++j)
#pragma unroll
      for (int i = 0; i < 8; ++i)
        sH[(rbase + i) * 64 + j * 16 + col] = (_Float16)hfrag[j][i];

    asm volatile("s_wait_dscnt 0x0" ::: "memory");

    // coalesced LDS -> global (f16) copy of this step's h for layer 1
    {
      int r2   = lane >> 1;
      int hoff = (lane & 1) * 32;
      const v8h* src = (const v8h*)(sH + r2 * 64 + hoff);
      v8h* dst = (v8h*)(h1 + ((size_t)(row0 + r2) * T_SEQ + t) * 64 + hoff);
#pragma unroll
      for (int q = 0; q < 4; ++q) dst[q] = src[q];
    }
  }
}

__global__ __launch_bounds__(32)
void gru_layer1(const _Float16* __restrict__ h1,    // (B, T, 64) f16
                const float* __restrict__ W_ih,     // (192, 64)
                const float* __restrict__ W_hh,     // (192, 64)
                const float* __restrict__ b_ih,
                const float* __restrict__ b_hh,
                const float* __restrict__ fc_W,     // (1, 64)
                const float* __restrict__ fc_b,     // (1,)
                float* __restrict__ out)            // (B,)
{
  __shared__ _Float16 sH[16 * 64];

  const int lane = threadIdx.x & 31;
  const int row0 = blockIdx.x * ROWS_PER_WAVE;
  const int col  = lane & 15;

  v16h Bih0[12], Bih1[12], Bhh0[12], Bhh1[12];
#pragma unroll
  for (int j = 0; j < 12; ++j) {
    Bih0[j] = load_bf32<64>(W_ih, j, 0, lane);
    Bih1[j] = load_bf32<64>(W_ih, j, 1, lane);
    Bhh0[j] = load_bf32<64>(W_hh, j, 0, lane);
    Bhh1[j] = load_bf32<64>(W_hh, j, 1, lane);
  }

  float hbrz[8], bxn[4], bhn[4];
#pragma unroll
  for (int j = 0; j < 8; ++j)
    hbrz[j] = 0.5f * (b_ih[j * 16 + col] + b_hh[j * 16 + col]);
#pragma unroll
  for (int j = 0; j < 4; ++j) {
    bxn[j] = b_ih[128 + j * 16 + col];
    bhn[j] = b_hh[128 + j * 16 + col];
  }

  for (int i = lane; i < 16 * 64; i += 32) sH[i] = (_Float16)0.0f;
  asm volatile("s_wait_dscnt 0x0" ::: "memory");

  v8f hfrag[4];
#pragma unroll
  for (int j = 0; j < 4; ++j) hfrag[j] = {};

  v16h xa0_n = load_ga(h1, row0, 0, 0, lane);
  v16h xa1_n = load_ga(h1, row0, 0, 1, lane);

  for (int t = 0; t < T_SEQ; ++t) {
    v16h xa0 = xa0_n, xa1 = xa1_n;
    int tn = (t + 1 < T_SEQ) ? t + 1 : t;
    xa0_n = load_ga(h1, row0, tn, 0, lane);
    xa1_n = load_ga(h1, row0, tn, 1, lane);

    v16h ha0 = load_a64(sH, 0, lane);
    v16h ha1 = load_a64(sH, 1, lane);

    v8f rz[8], xn[4], hn[4];
#pragma unroll
    for (int j = 0; j < 8; ++j) {
      v8f acc = {};
      acc = wmma(xa0, Bih0[j], acc);
      acc = wmma(xa1, Bih1[j], acc);
      acc = wmma(ha0, Bhh0[j], acc);
      acc = wmma(ha1, Bhh1[j], acc);
      rz[j] = acc;
    }
#pragma unroll
    for (int j = 0; j < 4; ++j) {
      v8f a1 = {}, a2 = {};
      a1 = wmma(xa0, Bih0[8 + j], a1);
      a1 = wmma(xa1, Bih1[8 + j], a1);
      a2 = wmma(ha0, Bhh0[8 + j], a2);
      a2 = wmma(ha1, Bhh1[8 + j], a2);
      xn[j] = a1; hn[j] = a2;
    }

#pragma unroll
    for (int j = 0; j < 4; ++j) {
#pragma unroll
      for (int i = 0; i < 8; ++i) {
        float rv = fmaf(0.5f, fast_tanh(fmaf(0.5f, rz[j][i], hbrz[j])), 0.5f);
        float zv = fmaf(0.5f, fast_tanh(fmaf(0.5f, rz[4 + j][i], hbrz[4 + j])), 0.5f);
        float na = xn[j][i] + bxn[j];
        na = fmaf(rv, hn[j][i], na);
        na = fmaf(rv, bhn[j], na);
        float nv = fast_tanh(na);
        hfrag[j][i] = fmaf(zv, hfrag[j][i] - nv, nv);
      }
    }

    const int rbase = (lane & 16) ? 8 : 0;
#pragma unroll
    for (int j = 0; j < 4; ++j)
#pragma unroll
      for (int i = 0; i < 8; ++i)
        sH[(rbase + i) * 64 + j * 16 + col] = (_Float16)hfrag[j][i];

    asm volatile("s_wait_dscnt 0x0" ::: "memory");
  }

  // FC: out[m] = sum_c h2_last[m,c] * fc_W[c] + fc_b
  float fcw[4];
#pragma unroll
  for (int j = 0; j < 4; ++j) fcw[j] = fc_W[j * 16 + col];
  const float fcb = fc_b[0];

  float part[8];
#pragma unroll
  for (int i = 0; i < 8; ++i)
    part[i] = hfrag[0][i] * fcw[0] + hfrag[1][i] * fcw[1] +
              hfrag[2][i] * fcw[2] + hfrag[3][i] * fcw[3];

#pragma unroll
  for (int m = 1; m <= 8; m <<= 1)
#pragma unroll
    for (int i = 0; i < 8; ++i)
      part[i] += __shfl_xor(part[i], m, 32);

  if ((lane & 15) == 0) {
    const int rb = (lane & 16) ? 8 : 0;
#pragma unroll
    for (int i = 0; i < 8; ++i)
      out[row0 + rb + i] = part[i] + fcb;
  }
}

extern "C" void kernel_launch(void* const* d_in, const int* in_sizes, int n_in,
                              void* d_out, int out_size, void* d_ws, size_t ws_size,
                              hipStream_t stream) {
  const float* x     = (const float*)d_in[0];
  const float* W_ih0 = (const float*)d_in[1];
  const float* W_hh0 = (const float*)d_in[2];
  const float* b_ih0 = (const float*)d_in[3];
  const float* b_hh0 = (const float*)d_in[4];
  const float* W_ih1 = (const float*)d_in[5];
  const float* W_hh1 = (const float*)d_in[6];
  const float* b_ih1 = (const float*)d_in[7];
  const float* b_hh1 = (const float*)d_in[8];
  const float* fc_W  = (const float*)d_in[9];
  const float* fc_b  = (const float*)d_in[10];

  _Float16* h1 = (_Float16*)d_ws;   // (B, T, 64) f16 = 128 MiB

  const int wgs = BATCH / ROWS_PER_WAVE;   // 128 workgroups, 1 wave each
  gru_layer0<<<wgs, 32, 0, stream>>>(x, W_ih0, W_hh0, b_ih0, b_hh0, h1);
  gru_layer1<<<wgs, 32, 0, stream>>>(h1, W_ih1, W_hh1, b_ih1, b_hh1,
                                     fc_W, fc_b, (float*)d_out);
}
